// ModulatedDeformConv2d_16965120819342
// MI455X (gfx1250) — compile-verified, matrix operand order checked by reading, not run on hardware
//
#include <hip/hip_runtime.h>
#include <hip/hip_bf16.h>

// ---- problem constants (from reference) ----
#define BATCH 4
#define CIN 64
#define HIMG 128
#define WIMG 128
#define HW (HIMG * WIMG)        // 16384
#define KTAPS 9                 // 3x3
#define OCH 64
#define KCOL (KTAPS * CIN)      // 576 reduction dim
#define KSTEPS (KCOL / 32)      // 18 WMMA K-steps

typedef __attribute__((ext_vector_type(16))) _Float16 v16h;
typedef __attribute__((ext_vector_type(8)))  _Float16 v8h;
typedef __attribute__((ext_vector_type(8)))  float    v8f;

// -----------------------------------------------------------------------------
// Kernel 1: repack weight (O,C,3,3) fp32 -> f16 A-fragment layout.
// Fragment storage: wf[((s*4 + otile)*32 + lane)*16 + j]
//   lane: m = lane&15 (row within o-tile), h = lane>>4
//   j<8  : k_local = 8h + j ; j>=8 : k_local = 16 + 8h + (j-8)
//   global k = s*32 + k_local ; k = ktap*64 + c (matches im2col column order)
// -----------------------------------------------------------------------------
__global__ void mdcn_wrepack(const float* __restrict__ weight,
                             _Float16* __restrict__ wf) {
  int idx = blockIdx.x * 256 + threadIdx.x;          // 0 .. 36863
  int j = idx & 15;
  int l = (idx >> 4) & 31;
  int w = (idx >> 9) & 3;                            // o-tile
  int s = idx >> 11;                                 // K-step
  int h = l >> 4;
  int m = l & 15;
  int kl = (j < 8) ? (8 * h + j) : (16 + 8 * h + (j - 8));
  int colk = s * 32 + kl;                            // 0..575
  int ktap = colk >> 6;                              // /64 -> tap
  int c = colk & 63;                                 // channel
  int o = w * 16 + m;
  wf[idx] = (_Float16)weight[((size_t)o * CIN + c) * KTAPS + ktap];
}

// -----------------------------------------------------------------------------
// Kernel 2: modulated deformable im2col, fp32 -> f16.
// One thread per (b, tap, pixel); loops channels in groups of 8 so stores are
// guaranteed-aligned global_store_b128 (dst offset is a multiple of 128 B).
// col layout: col[(b*HW + p)*KCOL + ktap*64 + c]  (row-major, row = pixel)
// -----------------------------------------------------------------------------
__global__ void mdcn_im2col(const float* __restrict__ x,
                            const float* __restrict__ off,
                            const float* __restrict__ msk,
                            _Float16* __restrict__ col) {
  int tid = blockIdx.x * 256 + threadIdx.x;          // 0 .. 4*9*16384-1
  int p  = tid & (HW - 1);
  int bk = tid >> 14;
  int k  = bk % KTAPS;
  int b  = bk / KTAPS;
  int ho = p >> 7;
  int wo = p & (WIMG - 1);

  size_t obase = ((size_t)b * (2 * KTAPS) + 2 * k) * HW + p;
  float oy = off[obase];
  float ox = off[obase + HW];
  float mv = msk[((size_t)b * KTAPS + k) * HW + p];

  // base grid position + tap offset (stride 1, pad 1, dil 1) + learned offset
  float sy = (float)(ho - 1 + (k / 3)) + oy;
  float sx = (float)(wo - 1 + (k % 3)) + ox;

  float y0f = floorf(sy), x0f = floorf(sx);
  int y0 = (int)y0f, x0 = (int)x0f;
  int y1 = y0 + 1,  x1 = x0 + 1;
  float wy = sy - y0f, wx = sx - x0f;

  float w00 = (1.f - wy) * (1.f - wx);
  float w01 = (1.f - wy) * wx;
  float w10 = wy * (1.f - wx);
  float w11 = wy * wx;

  bool vy0 = (y0 >= 0) & (y0 < HIMG);
  bool vy1 = (y1 >= 0) & (y1 < HIMG);
  bool vx0 = (x0 >= 0) & (x0 < WIMG);
  bool vx1 = (x1 >= 0) & (x1 < WIMG);
  if (!(vy0 & vx0)) w00 = 0.f;
  if (!(vy0 & vx1)) w01 = 0.f;
  if (!(vy1 & vx0)) w10 = 0.f;
  if (!(vy1 & vx1)) w11 = 0.f;

  int cy0 = min(max(y0, 0), HIMG - 1);
  int cy1 = min(max(y1, 0), HIMG - 1);
  int cx0 = min(max(x0, 0), WIMG - 1);
  int cx1 = min(max(x1, 0), WIMG - 1);

  int i00 = cy0 * WIMG + cx0;
  int i01 = cy0 * WIMG + cx1;
  int i10 = cy1 * WIMG + cx0;
  int i11 = cy1 * WIMG + cx1;

  const float* xb = x + (size_t)b * CIN * HW;
  _Float16* dst = col + (size_t)(b * HW + p) * KCOL + k * CIN;

  #pragma unroll
  for (int c8 = 0; c8 < CIN / 8; ++c8) {
    v8h pack;
    #pragma unroll
    for (int u = 0; u < 8; ++u) {
      const float* xc = xb + (size_t)(c8 * 8 + u) * HW;
      float v = w00 * xc[i00] + w01 * xc[i01] + w10 * xc[i10] + w11 * xc[i11];
      pack[u] = (_Float16)(v * mv);
    }
    *(v8h*)(dst + c8 * 8) = pack;                    // global_store_b128
  }
}

// -----------------------------------------------------------------------------
// Kernel 3: GEMM via v_wmma_f32_16x16x32_f16.
// Per block: 64 pixels x 64 out-channels (4 M-tiles x 4 N-tiles), 4 waves.
// Wave w owns o-tile w and iterates the 4 pixel tiles -> every col-buffer byte
// is loaded exactly once grid-wide; each A fragment feeds 4 WMMAs.
// B fragment per lane: pixel n = lane&15, h = lane>>4; two contiguous b128
// loads cover k_local {8h..8h+7} and {16+8h..16+8h+7}; pixel-tile t adds a
// compile-time 18432 B instruction offset.
// D layout: VGPR r -> o = otile*16 + r + 8h, pixel = p0 + t*16 + n  =>
// coalesced 64 B output stores along the pixel dimension.
// -----------------------------------------------------------------------------
__global__ void __launch_bounds__(128) mdcn_gemm(const _Float16* __restrict__ col,
                                                 const _Float16* __restrict__ wf,
                                                 const float* __restrict__ bias,
                                                 float* __restrict__ out) {
  int wave = threadIdx.x >> 5;
  int lane = threadIdx.x & 31;
  int h = lane >> 4;
  int n = lane & 15;

  int row_base = blockIdx.x * 64;                    // global pixel-row base
  int b  = row_base >> 14;                           // / HW (uniform per block)
  int p0 = row_base & (HW - 1);

  const _Float16* bp = col + (size_t)(row_base + n) * KCOL + 8 * h;
  const _Float16* ap = wf + (size_t)(wave * 32 + lane) * 16;

  v8f acc[4] = {};
  #pragma unroll
  for (int s = 0; s < KSTEPS; ++s) {
    v16h a = *(const v16h*)(ap + (size_t)s * (4 * 32 * 16));
    if (s + 1 < KSTEPS)
      __builtin_prefetch(bp + (s + 1) * 32, 0, 1);   // global_prefetch_b8
    #pragma unroll
    for (int t = 0; t < 4; ++t) {
      const _Float16* bt = bp + t * (16 * KCOL);     // +18432 B per pixel tile
      v8h b0 = *(const v8h*)(bt + s * 32);
      v8h b1 = *(const v8h*)(bt + s * 32 + 16);
      v16h bb = __builtin_shufflevector(b0, b1, 0, 1, 2, 3, 4, 5, 6, 7,
                                                8, 9, 10, 11, 12, 13, 14, 15);
      acc[t] = __builtin_amdgcn_wmma_f32_16x16x32_f16(
          /*neg_a=*/false, a, /*neg_b=*/false, bb,
          /*c_mod=*/(short)0, acc[t], /*reuse_a=*/false, /*reuse_b=*/false);
    }
  }

  #pragma unroll
  for (int t = 0; t < 4; ++t) {
    float* ob = out + ((size_t)b * OCH + wave * 16) * HW + p0 + t * 16 + n;
    #pragma unroll
    for (int r = 0; r < 8; ++r) {
      int o = r + 8 * h;                             // row within o-tile
      ob[(size_t)o * HW] = acc[t][r] + bias[wave * 16 + o];
    }
  }
}

// -----------------------------------------------------------------------------
extern "C" void kernel_launch(void* const* d_in, const int* in_sizes, int n_in,
                              void* d_out, int out_size, void* d_ws, size_t ws_size,
                              hipStream_t stream) {
  const float* x      = (const float*)d_in[0];
  const float* offset = (const float*)d_in[1];
  const float* mask   = (const float*)d_in[2];
  const float* weight = (const float*)d_in[3];
  const float* bias   = (const float*)d_in[4];
  float* out = (float*)d_out;

  // workspace: [ weight frags : 36864 f16 = 73728 B ][ col buffer : 65536*576 f16 ]
  _Float16* wf  = (_Float16*)d_ws;
  _Float16* col = (_Float16*)((char*)d_ws + 36864 * sizeof(_Float16));

  mdcn_wrepack<<<36864 / 256, 256, 0, stream>>>(weight, wf);
  mdcn_im2col<<<(BATCH * KTAPS * HW) / 256, 256, 0, stream>>>(x, offset, mask, col);
  mdcn_gemm<<<(BATCH * HW) / 64, 128, 0, stream>>>(col, wf, bias, out);
}
